// IPAttnProcessor_Self_13761075216749
// MI455X (gfx1250) — matrix operand order repaired and checked by reading, not compiled, not run on hardware
//
#include <hip/hip_runtime.h>

// ---------------------------------------------------------------------------
// IP-Adapter attention for MI455X (gfx1250, wave32, WMMA).
// All matmuls run on v_wmma_f32_16x16x32_bf16 (fp32 accumulate).
// K-chunk / GEMM A-tile staging uses GLOBAL_LOAD_ASYNC_TO_LDS_B128 when the
// toolchain exposes the builtin (ASYNCcnt-tracked, VGPR-bypassing copy).
// ---------------------------------------------------------------------------

typedef __bf16 bf16;
typedef __bf16 v16bf __attribute__((ext_vector_type(16)));
typedef float  v8f   __attribute__((ext_vector_type(8)));
typedef int    v4i   __attribute__((vector_size(16)));   // matches builtin param

#define B_   2
#define S_   2304
#define C_   1280
#define NH_  8
#define DH_  160
#define BH_  (B_ * NH_)
#define MH_  64
#define SR_  48
#define NEG_MAX (-3.402823466e38f)
#define INV_SQRT_DH 0.07905694150420949f   // 1/sqrt(160)

#define AS1 __attribute__((address_space(1)))
#define AS3 __attribute__((address_space(3)))

#if __has_builtin(__builtin_amdgcn_global_load_async_to_lds_b128)
#define HAVE_ASYNC_LDS 1
#else
#define HAVE_ASYNC_LDS 0
#endif

union V16U { v16bf v; uint4 q[2]; bf16 h[16]; };

__device__ __forceinline__ v8f v8f_zero() {
  v8f z = {0.f, 0.f, 0.f, 0.f, 0.f, 0.f, 0.f, 0.f};
  return z;
}

__device__ __forceinline__ v8f wmma_bf16(v16bf a, v16bf b, v8f c) {
  // (neg_a, A, neg_b, B, c_mod, C, reuse_a, reuse_b)
  return __builtin_amdgcn_wmma_f32_16x16x32_bf16(false, a, false, b, (short)0, c,
                                                 false, false);
}

// 16-byte global -> LDS copy; async (ASYNCcnt) when available.
__device__ __forceinline__ void ldg_to_lds_b128(const bf16* g, bf16* l) {
#if HAVE_ASYNC_LDS
  __builtin_amdgcn_global_load_async_to_lds_b128((AS1 v4i*)g, (AS3 v4i*)l, 0, 0);
#else
  *(uint4*)l = *(const uint4*)g;
#endif
}

__device__ __forceinline__ void wait_async_lds() {
#if HAVE_ASYNC_LDS
#if __has_builtin(__builtin_amdgcn_s_wait_asynccnt)
  __builtin_amdgcn_s_wait_asynccnt(0);
#else
  asm volatile("s_wait_asynccnt 0x0" ::: "memory");
#endif
#endif
}

// ---------------------------------------------------------------------------
// fp32 -> bf16 conversion
// ---------------------------------------------------------------------------
__global__ void cvt_f32_bf16(const float* __restrict__ src, bf16* __restrict__ dst,
                             int n) {
  int i = blockIdx.x * blockDim.x + threadIdx.x;
  if (i < n) dst[i] = (bf16)src[i];
}

// ---------------------------------------------------------------------------
// Nearest-interpolated masks: add_mask[b][s] in {1, -FMAX}; mig[b][s] = ig*scale
// ---------------------------------------------------------------------------
__global__ void prep_masks(const unsigned char* __restrict__ mask_id,
                           const float* __restrict__ mask_ig,
                           const float* __restrict__ scale,
                           float* __restrict__ add_mask,
                           float* __restrict__ mig) {
  int i = blockIdx.x * blockDim.x + threadIdx.x;
  if (i >= B_ * S_) return;
  int b = i / S_, s = i % S_;
  int r = s / SR_, c = s % SR_;
  int ih = (r * MH_) / SR_, iw = (c * MH_) / SR_;
  int mi = b * MH_ * MH_ + ih * MH_ + iw;
  add_mask[i] = mask_id[mi] ? 1.0f : NEG_MAX;
  mig[i] = mask_ig[mi] * scale[0];
}

// ---------------------------------------------------------------------------
// QKV GEMM: X[4608,1280] @ W[1280,1280]; output scattered into [BH,S,DH] bf16.
// 128x128x32 tiles, 8 waves (2x4), per-wave 64x32 = 4x2 WMMA frags.
// ---------------------------------------------------------------------------
__global__ __launch_bounds__(256) void gemm_qkv(
    const bf16* __restrict__ X, const bf16* __restrict__ Wq,
    const bf16* __restrict__ Wk, const bf16* __restrict__ Wv,
    bf16* __restrict__ qo, bf16* __restrict__ ko, bf16* __restrict__ vo) {
  __shared__ bf16 As[128 * 40];   // [m][k], K padded 32->40
  __shared__ bf16 Bs[128 * 40];   // [n][k] (transposed), padded

  const bf16* W = (blockIdx.z == 0) ? Wq : (blockIdx.z == 1) ? Wk : Wv;
  bf16* Out = (blockIdx.z == 0) ? qo : (blockIdx.z == 1) ? ko : vo;

  int m0 = blockIdx.x * 128, n0 = blockIdx.y * 128;
  int tid = threadIdx.x, lane = tid & 31, wave = tid >> 5;
  int half = lane >> 4, ln = lane & 15;
  int wm = wave >> 2, wn = wave & 3;

  v8f acc[4][2];
#pragma unroll
  for (int i = 0; i < 4; ++i)
#pragma unroll
    for (int j = 0; j < 2; ++j) acc[i][j] = v8f_zero();

  for (int kt = 0; kt < C_ / 32; ++kt) {
    int k0 = kt * 32;
    {  // A tile: straight copy -> async LDS path
      int row = tid >> 1, koff = (tid & 1) * 16;
      const bf16* src = &X[(size_t)(m0 + row) * C_ + k0 + koff];
      bf16* dst = &As[row * 40 + koff];
      ldg_to_lds_b128(src, dst);
      ldg_to_lds_b128(src + 8, dst + 8);
    }
    {  // B tile transposed into [n][k] (needs VGPR round trip)
      int kk = tid >> 3, noff = (tid & 7) * 16;
      V16U u;
      const uint4* src = (const uint4*)&W[(size_t)(k0 + kk) * C_ + n0 + noff];
      u.q[0] = src[0]; u.q[1] = src[1];
#pragma unroll
      for (int i = 0; i < 16; ++i) Bs[(noff + i) * 40 + kk] = u.h[i];
    }
    wait_async_lds();
    __syncthreads();

    v16bf af[4], bfr[2];
#pragma unroll
    for (int fm = 0; fm < 4; ++fm) {   // A frag: 16 contiguous K at half*16
      int m = wm * 64 + fm * 16 + ln;
      V16U u;
      const uint4* p = (const uint4*)&As[m * 40 + half * 16];
      u.q[0] = p[0]; u.q[1] = p[1];
      af[fm] = u.v;
    }
#pragma unroll
    for (int fn = 0; fn < 2; ++fn) {   // B frag: 16 contiguous K at half*16
      int n = wn * 32 + fn * 16 + ln;
      V16U u;
      const uint4* p = (const uint4*)&Bs[n * 40 + half * 16];
      u.q[0] = p[0]; u.q[1] = p[1];
      bfr[fn] = u.v;
    }
#pragma unroll
    for (int fm = 0; fm < 4; ++fm)
#pragma unroll
      for (int fn = 0; fn < 2; ++fn)
        acc[fm][fn] = wmma_bf16(af[fm], bfr[fn], acc[fm][fn]);
    __syncthreads();
  }

#pragma unroll
  for (int fm = 0; fm < 4; ++fm)
#pragma unroll
    for (int fn = 0; fn < 2; ++fn)
#pragma unroll
      for (int r = 0; r < 8; ++r) {
        int row = m0 + wm * 64 + fm * 16 + half * 8 + r;
        int col = n0 + wn * 32 + fn * 16 + ln;
        int b = row / S_, s = row % S_;
        int h = col / DH_, d = col % DH_;
        Out[((size_t)(b * NH_ + h) * S_ + s) * DH_ + d] = (bf16)acc[fm][fn][r];
      }
}

// ---------------------------------------------------------------------------
// Output projection: comb[4608,1280] @ Wo + bo -> fp32 d_out
// ---------------------------------------------------------------------------
__global__ __launch_bounds__(256) void gemm_out(
    const bf16* __restrict__ X, const bf16* __restrict__ W,
    const float* __restrict__ bo, float* __restrict__ out) {
  __shared__ bf16 As[128 * 40];
  __shared__ bf16 Bs[128 * 40];

  int m0 = blockIdx.x * 128, n0 = blockIdx.y * 128;
  int tid = threadIdx.x, lane = tid & 31, wave = tid >> 5;
  int half = lane >> 4, ln = lane & 15;
  int wm = wave >> 2, wn = wave & 3;

  v8f acc[4][2];
#pragma unroll
  for (int i = 0; i < 4; ++i)
#pragma unroll
    for (int j = 0; j < 2; ++j) acc[i][j] = v8f_zero();

  for (int kt = 0; kt < C_ / 32; ++kt) {
    int k0 = kt * 32;
    {
      int row = tid >> 1, koff = (tid & 1) * 16;
      const bf16* src = &X[(size_t)(m0 + row) * C_ + k0 + koff];
      bf16* dst = &As[row * 40 + koff];
      ldg_to_lds_b128(src, dst);
      ldg_to_lds_b128(src + 8, dst + 8);
    }
    {
      int kk = tid >> 3, noff = (tid & 7) * 16;
      V16U u;
      const uint4* src = (const uint4*)&W[(size_t)(k0 + kk) * C_ + n0 + noff];
      u.q[0] = src[0]; u.q[1] = src[1];
#pragma unroll
      for (int i = 0; i < 16; ++i) Bs[(noff + i) * 40 + kk] = u.h[i];
    }
    wait_async_lds();
    __syncthreads();

    v16bf af[4], bfr[2];
#pragma unroll
    for (int fm = 0; fm < 4; ++fm) {
      int m = wm * 64 + fm * 16 + ln;
      V16U u;
      const uint4* p = (const uint4*)&As[m * 40 + half * 16];
      u.q[0] = p[0]; u.q[1] = p[1];
      af[fm] = u.v;
    }
#pragma unroll
    for (int fn = 0; fn < 2; ++fn) {
      int n = wn * 32 + fn * 16 + ln;
      V16U u;
      const uint4* p = (const uint4*)&Bs[n * 40 + half * 16];
      u.q[0] = p[0]; u.q[1] = p[1];
      bfr[fn] = u.v;
    }
#pragma unroll
    for (int fm = 0; fm < 4; ++fm)
#pragma unroll
      for (int fn = 0; fn < 2; ++fn)
        acc[fm][fn] = wmma_bf16(af[fm], bfr[fn], acc[fm][fn]);
    __syncthreads();
  }

#pragma unroll
  for (int fm = 0; fm < 4; ++fm)
#pragma unroll
    for (int fn = 0; fn < 2; ++fn)
#pragma unroll
      for (int r = 0; r < 8; ++r) {
        int row = m0 + wm * 64 + fm * 16 + half * 8 + r;
        int col = n0 + wn * 32 + fn * 16 + ln;
        out[(size_t)row * C_ + col] = acc[fm][fn][r] + bo[col];
      }
}

// ---------------------------------------------------------------------------
// Flash attention pass over one (bh, 16-query tile) per wave. K/V staged in
// LDS in 32-key chunks shared by all 8 waves of the block.
//   Ks: [key][d]  (async copy)  -> score B-frags are contiguous b128 reads
//   Vs: [d][key]  (transposed)  -> PV    B-frags are contiguous b128 reads
// ---------------------------------------------------------------------------
__device__ __forceinline__ void attn_pass(
    const bf16* __restrict__ kp, const bf16* __restrict__ vp,
    const float* __restrict__ maskrow,   // per-key additive mask or nullptr
    int bh, int half, int ln, int tid,
    const v16bf (&qf)[5],
    bf16* Ks, bf16* Vs, bf16* Pw,
    v8f (&o)[10]) {
  float ml[8], ll[8];
#pragma unroll
  for (int f = 0; f < 10; ++f) o[f] = v8f_zero();
#pragma unroll
  for (int r = 0; r < 8; ++r) { ml[r] = NEG_MAX; ll[r] = 0.f; }

  for (int chunk = 0; chunk < S_ / 32; ++chunk) {
    int key0 = chunk * 32;
    __syncthreads();
    for (int c = tid; c < 640; c += 256) {     // 32x160 bf16 in 8-elem chunks
      int e = c * 8;
      int key = e / DH_, d = e % DH_;
      size_t g = ((size_t)bh * S_ + key0 + key) * DH_ + d;
      // K: straight copy (async path)
      ldg_to_lds_b128(&kp[g], &Ks[key * 168 + d]);
      // V: transpose into [d][key] so PV fragments read contiguously
      V16U u;
      u.q[0] = *(const uint4*)&vp[g];
#pragma unroll
      for (int i = 0; i < 8; ++i) Vs[(d + i) * 40 + key] = u.h[i];
    }
    wait_async_lds();
    __syncthreads();

    // scores for two 16-key tiles (keys key0..key0+31)
    v8f sc0 = v8f_zero(), sc1 = v8f_zero();
#pragma unroll
    for (int t = 0; t < 5; ++t) {
      V16U u0, u1;
      const uint4* p0 = (const uint4*)&Ks[ln * 168 + t * 32 + half * 16];
      const uint4* p1 = (const uint4*)&Ks[(16 + ln) * 168 + t * 32 + half * 16];
      u0.q[0] = p0[0]; u0.q[1] = p0[1];
      u1.q[0] = p1[0]; u1.q[1] = p1[1];
      sc0 = wmma_bf16(qf[t], u0.v, sc0);
      sc1 = wmma_bf16(qf[t], u1.v, sc1);
    }

    float mk0 = 0.f, mk1 = 0.f;
    if (maskrow) {
      mk0 = maskrow[key0 + ln];
      mk1 = maskrow[key0 + 16 + ln];
    }

    float alpha[8];
#pragma unroll
    for (int r = 0; r < 8; ++r) {
      float s0 = sc0[r] * INV_SQRT_DH + mk0;
      float s1 = sc1[r] * INV_SQRT_DH + mk1;
      float mx = fmaxf(s0, s1);
      mx = fmaxf(mx, __shfl_xor(mx, 1));
      mx = fmaxf(mx, __shfl_xor(mx, 2));
      mx = fmaxf(mx, __shfl_xor(mx, 4));
      mx = fmaxf(mx, __shfl_xor(mx, 8));
      float mnew = fmaxf(ml[r], mx);
      float a = __expf(ml[r] - mnew);
      ml[r] = mnew;
      alpha[r] = a;
      float p0 = __expf(s0 - mnew);
      float p1 = __expf(s1 - mnew);
      float rs = p0 + p1;
      rs += __shfl_xor(rs, 1);
      rs += __shfl_xor(rs, 2);
      rs += __shfl_xor(rs, 4);
      rs += __shfl_xor(rs, 8);
      ll[r] = ll[r] * a + rs;
      Pw[(half * 8 + r) * 40 + ln]      = (bf16)p0;
      Pw[(half * 8 + r) * 40 + 16 + ln] = (bf16)p1;
    }

#pragma unroll
    for (int f = 0; f < 10; ++f)
#pragma unroll
      for (int r = 0; r < 8; ++r) o[f][r] *= alpha[r];

    // P (16x32) A-fragment from per-wave LDS (same-wave LDS is in-order)
    V16U ua;
    const uint4* pa = (const uint4*)&Pw[ln * 40 + half * 16];
    ua.q[0] = pa[0]; ua.q[1] = pa[1];

#pragma unroll
    for (int fn = 0; fn < 10; ++fn) {
      V16U ub;
      const uint4* pv = (const uint4*)&Vs[(fn * 16 + ln) * 40 + half * 16];
      ub.q[0] = pv[0]; ub.q[1] = pv[1];
      o[fn] = wmma_bf16(ua.v, ub.v, o[fn]);
    }
  }

#pragma unroll
  for (int f = 0; f < 10; ++f)
#pragma unroll
    for (int r = 0; r < 8; ++r) o[f][r] /= ll[r];
}

__global__ __launch_bounds__(256) void attn_kernel(
    const bf16* __restrict__ q, const bf16* __restrict__ k,
    const bf16* __restrict__ v, const bf16* __restrict__ ks,
    const bf16* __restrict__ vs, const float* __restrict__ add_mask,
    const float* __restrict__ mig, bf16* __restrict__ comb) {
  __shared__ bf16 Ks[32 * 168];    // [key][d], DH padded 160->168
  __shared__ bf16 Vs[160 * 40];    // [d][key], keys padded 32->40
  __shared__ bf16 Pb[8][16 * 40];  // per-wave P tile, padded 32->40

  int tid = threadIdx.x, lane = tid & 31, wave = tid >> 5;
  int half = lane >> 4, ln = lane & 15;
  int bh = blockIdx.x;
  int b = bh >> 3, h = bh & 7;
  int maskb = bh & 1;  // jnp.tile((NH,1,1)) quirk: mask batch = bh % B
  int qrow0 = (blockIdx.y * 8 + wave) * 16;
  bf16* Pw = Pb[wave];

  // Q tile 16xDH as 5 A-fragments (kept in VGPRs for both passes)
  v16bf qf[5];
#pragma unroll
  for (int t = 0; t < 5; ++t) {
    const bf16* p = q + ((size_t)bh * S_ + qrow0 + ln) * DH_ + t * 32 + half * 16;
    V16U u;
    u.q[0] = *(const uint4*)p;
    u.q[1] = *(const uint4*)(p + 8);
    qf[t] = u.v;
  }

  v8f o_self[10], o_ip[10];
  attn_pass(k, v, (const float*)nullptr, bh, half, ln, tid, qf, Ks, Vs, Pw, o_self);
  attn_pass(ks, vs, add_mask + (size_t)maskb * S_, bh, half, ln, tid, qf, Ks, Vs, Pw, o_ip);

  float migv[8];
#pragma unroll
  for (int r = 0; r < 8; ++r) migv[r] = mig[(size_t)b * S_ + qrow0 + half * 8 + r];

#pragma unroll
  for (int fn = 0; fn < 10; ++fn)
#pragma unroll
    for (int r = 0; r < 8; ++r) {
      int srow = qrow0 + half * 8 + r;
      int d = fn * 16 + ln;
      comb[((size_t)b * S_ + srow) * C_ + h * DH_ + d] =
          (bf16)(o_self[fn][r] + o_ip[fn][r] * migv[r]);
    }
}

// ---------------------------------------------------------------------------
// Host launch
// ---------------------------------------------------------------------------
extern "C" void kernel_launch(void* const* d_in, const int* in_sizes, int n_in,
                              void* d_out, int out_size, void* d_ws, size_t ws_size,
                              hipStream_t stream) {
  (void)in_sizes; (void)n_in; (void)out_size; (void)ws_size;
  const float* hs  = (const float*)d_in[0];
  const float* Wq  = (const float*)d_in[1];
  const float* Wk  = (const float*)d_in[2];
  const float* Wv  = (const float*)d_in[3];
  const float* Wo  = (const float*)d_in[4];
  const float* bo  = (const float*)d_in[5];
  const float* sks = (const float*)d_in[6];
  const float* svs = (const float*)d_in[7];
  const unsigned char* mask_id = (const unsigned char*)d_in[8];
  const float* mask_ig = (const float*)d_in[9];
  const float* scale   = (const float*)d_in[10];
  float* out = (float*)d_out;

  char* ws = (char*)d_ws;
  size_t off = 0;
  auto alloc = [&](size_t bytes) -> void* {
    void* p = ws + off;
    off = (off + bytes + 255) & ~(size_t)255;
    return p;
  };
  const size_t NXC  = (size_t)B_ * S_ * C_;          // 5,898,240
  const size_t NCC  = (size_t)C_ * C_;               // 1,638,400
  const size_t NHSD = (size_t)BH_ * S_ * DH_;        // 5,898,240

  bf16* Xb   = (bf16*)alloc(NXC * 2);
  bf16* Wqb  = (bf16*)alloc(NCC * 2);
  bf16* Wkb  = (bf16*)alloc(NCC * 2);
  bf16* Wvb  = (bf16*)alloc(NCC * 2);
  bf16* Wob  = (bf16*)alloc(NCC * 2);
  bf16* qb   = (bf16*)alloc(NHSD * 2);
  bf16* kb   = (bf16*)alloc(NHSD * 2);
  bf16* vb   = (bf16*)alloc(NHSD * 2);
  bf16* ksb  = (bf16*)alloc(NHSD * 2);
  bf16* vsb  = (bf16*)alloc(NHSD * 2);
  bf16* combb = (bf16*)alloc(NXC * 2);
  float* amask = (float*)alloc((size_t)B_ * S_ * 4);
  float* migv  = (float*)alloc((size_t)B_ * S_ * 4);

  auto cvt = [&](const float* s, bf16* d, size_t n) {
    cvt_f32_bf16<<<(unsigned)((n + 255) / 256), 256, 0, stream>>>(s, d, (int)n);
  };
  cvt(hs, Xb, NXC);
  cvt(Wq, Wqb, NCC);
  cvt(Wk, Wkb, NCC);
  cvt(Wv, Wvb, NCC);
  cvt(Wo, Wob, NCC);
  cvt(sks, ksb, NHSD);
  cvt(svs, vsb, NHSD);

  prep_masks<<<(B_ * S_ + 255) / 256, 256, 0, stream>>>(mask_id, mask_ig, scale,
                                                        amask, migv);

  gemm_qkv<<<dim3((B_ * S_) / 128, C_ / 128, 3), 256, 0, stream>>>(
      Xb, Wqb, Wkb, Wvb, qb, kb, vb);

  attn_kernel<<<dim3(BH_, (S_ / 16) / 8), 256, 0, stream>>>(
      qb, kb, vb, ksb, vsb, amask, migv, combb);

  gemm_out<<<dim3((B_ * S_) / 128, C_ / 128), 256, 0, stream>>>(combb, Wob, bo, out);
}
